// ConvLSTM_481036337485
// MI455X (gfx1250) — compile-verified, matrix operand order checked by reading, not run on hardware
//
#include <hip/hip_runtime.h>
#include <hip/hip_bf16.h>
#include <math.h>

// ---------------------------------------------------------------------------
// ConvLSTM (B=32, T=16, H=64, W=64, Cin=3, F=8) on gfx1250 via f16 WMMA.
// Fused conv(x,Wx)+conv(h,Wh) as one im2col GEMM: K = (tap-pair, cin16) = 32,
// exactly one v_wmma_f32_16x16x32_f16 per (tap-pair, 16-cout tile).
// ---------------------------------------------------------------------------

typedef __attribute__((ext_vector_type(16))) _Float16 v16h;
typedef __attribute__((ext_vector_type(8)))  _Float16 v8h;
typedef __attribute__((ext_vector_type(8)))  float    v8f;

#define B_    32
#define T_    16
#define H_    64
#define W_    64
#define CIN_  3
#define F_    8
#define COUT_ 32            // 4*F
#define CPAD_ 16            // cin padded: 0..2 = x, 3..10 = h, 11..15 = 0
#define NPAIR_ 5            // 9 taps -> 4 pairs + 1 half pair
#define HCN_  (B_*H_*W_*F_) // elements in h / c buffers (1,048,576)

// ---------------------------------------------------------------------------
// Pack [W_x | W_h | zeros] into B-fragment-friendly layout:
//   Wp[pair][ntile][n][kk]  (f16),  kk = 0..31 ; tap = pair*2 + (kk>>4),
//   cin = kk&15, cout = ntile*16 + n.  tap 9 (half of pair 4) -> zeros.
// ---------------------------------------------------------------------------
__global__ void clstm_pack_w(const float* __restrict__ Wx,
                             const float* __restrict__ Wh,
                             _Float16* __restrict__ Wp) {
    int idx = blockIdx.x * blockDim.x + threadIdx.x;      // 5*2*16*32 = 5120
    if (idx >= NPAIR_ * 2 * 16 * 32) return;
    int kk    = idx & 31;
    int n     = (idx >> 5) & 15;
    int ntile = (idx >> 9) & 1;
    int pair  = idx >> 10;
    int tap   = pair * 2 + (kk >> 4);
    int cin   = kk & 15;
    int cout  = ntile * 16 + n;
    float v = 0.0f;
    if (tap < 9) {
        if (cin < CIN_)            v = Wx[(tap * CIN_ + cin) * COUT_ + cout];
        else if (cin < CIN_ + F_)  v = Wh[(tap * F_ + (cin - CIN_)) * COUT_ + cout];
    }
    Wp[idx] = (_Float16)v;
}

__global__ void clstm_zero(float* __restrict__ a, float* __restrict__ b, int n) {
    int i = blockIdx.x * blockDim.x + threadIdx.x;
    if (i < n) { a[i] = 0.0f; b[i] = 0.0f; }
}

__device__ __forceinline__ float hsig(float x) {
    return fminf(fmaxf(0.2f * x + 0.5f, 0.0f), 1.0f);
}

// ---------------------------------------------------------------------------
// One timestep. Grid = B*H blocks (one block per (batch, image row)),
// 128 threads = 4 waves, each wave owns 16 output pixels.
// ---------------------------------------------------------------------------
__global__ void __launch_bounds__(128)
clstm_step(const float* __restrict__ x, int t,
           const float* __restrict__ h_in, float* __restrict__ h_out,
           float* __restrict__ cbuf,
           const _Float16* __restrict__ Wp,
           const float* __restrict__ bias) {
    // Halo'd combined-channel patch: 3 rows x 66 cols x 16 cin (f16) = 6336 B
    __shared__ __align__(16) _Float16 comb[3][66][CPAD_];
    // Packed weights: 5 pairs x 2 ntiles x 16 n x 32 kk (f16) = 10240 B
    __shared__ __align__(16) _Float16 wlds[NPAIR_][2][16][32];
    // Gate pre-activations: 64 px x 32 cout (f32) = 8192 B
    __shared__ float zbuf[W_][COUT_];

    const int tid = threadIdx.x;
    const int b   = blockIdx.x >> 6;   // batch
    const int y   = blockIdx.x & 63;   // image row

    // ---- stage packed weights to LDS (2560 dwords) ----
    {
        const uint32_t* src = (const uint32_t*)Wp;
        uint32_t* dst = (uint32_t*)&wlds[0][0][0][0];
        for (int i = tid; i < NPAIR_ * 2 * 16 * 32 / 2; i += 128) dst[i] = src[i];
    }

    // ---- stage halo'd (x ++ h) patch, f32 -> f16 ----
    for (int site = tid; site < 3 * 66; site += 128) {
        int r  = site / 66;            // patch row 0..2  (y + r - 1)
        int px = site - r * 66;        // patch col 0..65 (x = px - 1)
        int yy = y + r - 1;
        int xx = px - 1;
        bool valid = (yy >= 0) & (yy < H_) & (xx >= 0) & (xx < W_);
        float xv0 = 0.f, xv1 = 0.f, xv2 = 0.f;
        float hv[F_] = {0.f, 0.f, 0.f, 0.f, 0.f, 0.f, 0.f, 0.f};
        if (valid) {
            const float* xp = x + ((size_t)((b * T_ + t) * H_ + yy) * W_ + xx) * CIN_;
            xv0 = xp[0]; xv1 = xp[1]; xv2 = xp[2];
            const float* hp = h_in + ((size_t)(b * H_ + yy) * W_ + xx) * F_;
#pragma unroll
            for (int c = 0; c < F_; ++c) hv[c] = hp[c];
        }
        _Float16* dst = &comb[r][px][0];
        dst[0] = (_Float16)xv0; dst[1] = (_Float16)xv1; dst[2] = (_Float16)xv2;
#pragma unroll
        for (int c = 0; c < F_; ++c) dst[3 + c] = (_Float16)hv[c];
#pragma unroll
        for (int c = CIN_ + F_; c < CPAD_; ++c) dst[c] = (_Float16)0.0f;
    }
    __syncthreads();

    // ---- WMMA: 16 px x 32 cout per wave, K = two taps x 16 cin ----
    const int wv   = tid >> 5;         // wave 0..3 -> pixel segment
    const int lane = tid & 31;
    const int m    = lane & 15;        // A row (pixel) / B column (cout)
    const int kh   = lane >> 4;        // lane half selects K octet

    v8f acc0 = {};                     // couts 0..15
    v8f acc1 = {};                     // couts 16..31
    const int colbase = wv * 16 + m;   // patch column of this lane's pixel

#pragma unroll
    for (int pair = 0; pair < NPAIR_; ++pair) {
        const int tA = pair * 2, tB = tA + 1;
        const int dyA = tA / 3, dxA = tA - dyA * 3;
        v8h lo = *(const v8h*)&comb[dyA][colbase + dxA][kh * 8];
        v8h hi = {};
        if (tB < 9) {
            const int dyB = tB / 3, dxB = tB - dyB * 3;
            hi = *(const v8h*)&comb[dyB][colbase + dxB][kh * 8];
        }
        v16h a = __builtin_shufflevector(lo, hi, 0, 1, 2, 3, 4, 5, 6, 7,
                                                 8, 9, 10, 11, 12, 13, 14, 15);
        v8h b0l = *(const v8h*)&wlds[pair][0][m][kh * 8];
        v8h b0h = *(const v8h*)&wlds[pair][0][m][16 + kh * 8];
        v8h b1l = *(const v8h*)&wlds[pair][1][m][kh * 8];
        v8h b1h = *(const v8h*)&wlds[pair][1][m][16 + kh * 8];
        v16h bf0 = __builtin_shufflevector(b0l, b0h, 0, 1, 2, 3, 4, 5, 6, 7,
                                                     8, 9, 10, 11, 12, 13, 14, 15);
        v16h bf1 = __builtin_shufflevector(b1l, b1h, 0, 1, 2, 3, 4, 5, 6, 7,
                                                     8, 9, 10, 11, 12, 13, 14, 15);
        acc0 = __builtin_amdgcn_wmma_f32_16x16x32_f16(false, a, false, bf0,
                                                      (short)0, acc0, false, false);
        acc1 = __builtin_amdgcn_wmma_f32_16x16x32_f16(false, a, false, bf1,
                                                      (short)0, acc1, false, false);
    }

    // ---- spill accumulators: C/D layout VGPR r -> M = r + 8*kh, N = m ----
#pragma unroll
    for (int r = 0; r < 8; ++r) {
        int px = wv * 16 + r + 8 * kh;
        zbuf[px][m]      = acc0[r];
        zbuf[px][16 + m] = acc1[r];
    }
    __syncthreads();

    // ---- pointwise LSTM gates (keras order i,f,c,o), f32 ----
    for (int q = tid; q < W_ * F_; q += 128) {
        int px = q >> 3;
        int f  = q & 7;
        float zi = zbuf[px][f]      + bias[f];
        float zf = zbuf[px][f + 8]  + bias[f + 8];
        float zc = zbuf[px][f + 16] + bias[f + 16];
        float zo = zbuf[px][f + 24] + bias[f + 24];
        size_t idx = (size_t)(blockIdx.x * W_ + px) * F_ + f;
        float cp = cbuf[idx];
        float ig = hsig(zi);
        float fg = hsig(zf);
        float cn = fg * cp + ig * tanhf(zc);
        float og = hsig(zo);
        cbuf[idx]  = cn;
        h_out[idx] = og * tanhf(cn);
    }
}

// ---------------------------------------------------------------------------
// Head: logits[b] = h_final(b,:) @ W_out + b_out ; softmax. One block / batch.
// ---------------------------------------------------------------------------
__global__ void __launch_bounds__(256)
clstm_head(const float* __restrict__ h, const float* __restrict__ Wout,
           const float* __restrict__ bout, float* __restrict__ out) {
    __shared__ float red[256][4];
    const int b = blockIdx.x, t = threadIdx.x;
    const int K = H_ * W_ * F_;   // 32768
    float acc0 = 0.f, acc1 = 0.f, acc2 = 0.f, acc3 = 0.f;
    const float* hb = h + (size_t)b * K;
    for (int k = t; k < K; k += 256) {
        float hv = hb[k];
        const float* wr = Wout + (size_t)k * 4;
        acc0 += hv * wr[0]; acc1 += hv * wr[1];
        acc2 += hv * wr[2]; acc3 += hv * wr[3];
    }
    red[t][0] = acc0; red[t][1] = acc1; red[t][2] = acc2; red[t][3] = acc3;
    __syncthreads();
    for (int s = 128; s > 0; s >>= 1) {
        if (t < s) {
#pragma unroll
            for (int j = 0; j < 4; ++j) red[t][j] += red[t + s][j];
        }
        __syncthreads();
    }
    if (t == 0) {
        float lg[4];
#pragma unroll
        for (int j = 0; j < 4; ++j) lg[j] = red[0][j] + bout[j];
        float mx = fmaxf(fmaxf(lg[0], lg[1]), fmaxf(lg[2], lg[3]));
        float e[4], s = 0.f;
#pragma unroll
        for (int j = 0; j < 4; ++j) { e[j] = expf(lg[j] - mx); s += e[j]; }
        float inv = 1.0f / s;
#pragma unroll
        for (int j = 0; j < 4; ++j) out[b * 4 + j] = e[j] * inv;
    }
}

// ---------------------------------------------------------------------------
extern "C" void kernel_launch(void* const* d_in, const int* in_sizes, int n_in,
                              void* d_out, int out_size, void* d_ws, size_t ws_size,
                              hipStream_t stream) {
    const float* x    = (const float*)d_in[0];
    const float* Wx   = (const float*)d_in[1];
    const float* Wh   = (const float*)d_in[2];
    const float* bias = (const float*)d_in[3];
    const float* Wout = (const float*)d_in[4];
    const float* bout = (const float*)d_in[5];
    float* out = (float*)d_out;

    char* ws = (char*)d_ws;
    _Float16* Wp = (_Float16*)ws;                 // 10240 B packed weights
    float* h0   = (float*)(ws + 16384);           // 4 MB
    float* h1   = h0 + HCN_;                      // 4 MB
    float* cbuf = h1 + HCN_;                      // 4 MB

    clstm_pack_w<<<(NPAIR_ * 2 * 16 * 32 + 255) / 256, 256, 0, stream>>>(Wx, Wh, Wp);
    clstm_zero<<<(HCN_ + 255) / 256, 256, 0, stream>>>(h0, cbuf, HCN_);

    for (int t = 0; t < T_; ++t) {
        const float* hin = (t & 1) ? h1 : h0;
        float*       hot = (t & 1) ? h0 : h1;
        clstm_step<<<B_ * H_, 128, 0, stream>>>(x, t, hin, hot, cbuf, Wp, bias);
    }
    // t=15 (odd) wrote h0 -> final hidden state is in h0
    clstm_head<<<B_, 256, 0, stream>>>(h0, Wout, bout, out);
}
